// GNNStack_25580825215361
// MI455X (gfx1250) — matrix operand order repaired and checked by reading, not compiled, not run on hardware
//
#include <hip/hip_runtime.h>

// ---------------------------------------------------------------------------
// GNNStack for MI455X (gfx1250).
//   - Weights pre-packed once per launch into wave32 WMMA B-fragment layout
//     (bf16, 32B/lane contiguous) -> inner loop B reads are 2x global_load_b128.
//   - Activations converted f32->bf16 in registers; f32 WMMA accumulation.
//   - Edge scatter-add with global_atomic_add_f32; both the 19.2MB message
//     table and the accumulator fit in MI455X's 192MB L2, so the random
//     gather/atomic traffic stays on-die. HBM only sees streaming GEMM I/O.
// ---------------------------------------------------------------------------

typedef __attribute__((ext_vector_type(16))) __bf16 v16bf;
typedef __attribute__((ext_vector_type(8)))  float  v8f;

// ---------------------------------------------------------------------------
// Pack W[K, NT*16] (f32 row-major) into WMMA B-fragment layout:
//   P[((kt*NT + nt)*32 + lane)*16 + e] = bf16( W[(kt*32 + half*16 + e)*ldn + col] )
// where half = lane>>4, col = nt*16 + (lane&15).  One thread per (kt,nt,lane).
// ---------------------------------------------------------------------------
__global__ void pack_w_bf16(const float* __restrict__ W, __bf16* __restrict__ P,
                            int K, int NT) {
    const int tid  = blockIdx.x * blockDim.x + threadIdx.x;
    const int ktiles = K >> 5;
    if (tid >= ktiles * NT * 32) return;
    const int lane = tid & 31;
    const int nt   = (tid >> 5) % NT;
    const int kt   = (tid >> 5) / NT;
    const int half = lane >> 4;
    const int col  = nt * 16 + (lane & 15);
    const int ldn  = NT * 16;
    __bf16* dst = P + ((size_t)(kt * NT + nt) * 32 + lane) * 16;
    const float* src = W + (size_t)(kt * 32 + half * 16) * ldn + col;
#pragma unroll
    for (int e = 0; e < 16; ++e) dst[e] = (__bf16)src[(size_t)e * ldn];
}

// A fragment (16x32 bf16): lane(m=l&15, half=l>>4) holds
// elems 0..15 = A[m, k0+half*16 .. +15].
// Two 16B-aligned float4 pairs -> global_load_b128 x4, then packed cvts.
__device__ __forceinline__ v16bf load_a_frag(const float* __restrict__ pa) {
    const float4* p = reinterpret_cast<const float4*>(pa);
    const float4* q = reinterpret_cast<const float4*>(pa + 8);
    float4 a0 = p[0], a1 = p[1], a2 = q[0], a3 = q[1];
    v16bf r;
    r[0]  = (__bf16)a0.x; r[1]  = (__bf16)a0.y; r[2]  = (__bf16)a0.z; r[3]  = (__bf16)a0.w;
    r[4]  = (__bf16)a1.x; r[5]  = (__bf16)a1.y; r[6]  = (__bf16)a1.z; r[7]  = (__bf16)a1.w;
    r[8]  = (__bf16)a2.x; r[9]  = (__bf16)a2.y; r[10] = (__bf16)a2.z; r[11] = (__bf16)a2.w;
    r[12] = (__bf16)a3.x; r[13] = (__bf16)a3.y; r[14] = (__bf16)a3.z; r[15] = (__bf16)a3.w;
    return r;
}

// C[M, NT*16] = A[M, K] @ W + bias (W pre-packed bf16 fragments), optional ReLU.
// One wave computes a 16-row x (NT*16)-col strip. Requires M%16==0, K%32==0.
template <int NT>
__global__ void gemm_packed(const float* __restrict__ A,
                            const __bf16* __restrict__ Bp,
                            const float* __restrict__ bias,
                            float* __restrict__ C,
                            int M, int K, int relu) {
    const int ldn  = NT * 16;
    const int lane = threadIdx.x & 31;
    const int wave = threadIdx.x >> 5;
    const int rt   = blockIdx.x * (blockDim.x >> 5) + wave;
    const int mtiles = M >> 4;
    if (rt >= mtiles) return;            // uniform per wave: whole wave exits

    const int half = lane >> 4;
    const int mloc = lane & 15;
    const long rowA = (long)(rt * 16 + mloc);
    const v16bf* Bf = reinterpret_cast<const v16bf*>(Bp);

    v8f acc[NT];
#pragma unroll
    for (int t = 0; t < NT; ++t)
#pragma unroll
        for (int r = 0; r < 8; ++r) acc[t][r] = 0.0f;

    for (int k0 = 0; k0 < K; k0 += 32) {
        const float* pa = A + rowA * K + k0 + half * 16;
        if (k0 + 32 < K) __builtin_prefetch(pa + 32, 0, 1);   // global_prefetch_b8
        const v16bf a = load_a_frag(pa);
        const size_t kb = (size_t)(k0 >> 5) * NT * 32 + lane;
#pragma unroll
        for (int t = 0; t < NT; ++t) {
            const v16bf b = Bf[kb + (size_t)t * 32];          // 2x global_load_b128
            acc[t] = __builtin_amdgcn_wmma_f32_16x16x32_bf16(
                false, a, false, b, (short)0, acc[t], false, false);
        }
    }

    // D layout: vgpr r <-> row (r + 8*half), col = lane&15
    const int n = lane & 15;
#pragma unroll
    for (int t = 0; t < NT; ++t) {
        const int col = t * 16 + n;
        const float bv = bias[col];
#pragma unroll
        for (int r = 0; r < 8; ++r) {
            float v = acc[t][r] + bv;
            if (relu) v = v > 0.0f ? v : 0.0f;
            C[(long)(rt * 16 + r + half * 8) * ldn + col] = v;
        }
    }
}

// Edge scatter: H[dst, :] += T[src, :] for src != dst. One thread per
// (edge, 4-feature group); rows are 384B so float4-aligned. L2-resident.
__global__ void scatter_edges(const float* __restrict__ T,
                              const long long* __restrict__ ei,
                              float* __restrict__ H, int E, int F) {
    const int groups = F >> 2;
    const long tid = (long)blockIdx.x * blockDim.x + threadIdx.x;
    const long e = tid / groups;
    const int  g = (int)(tid % groups);
    if (e >= E) return;
    const long long s = ei[e];
    const long long d = ei[(long)E + e];
    if (s == d) return;                       // remove_self_loops
    const float4 v = *reinterpret_cast<const float4*>(T + (long)s * F + g * 4);
    float* hp = H + (long)d * F + g * 4;
    atomicAdd(hp + 0, v.x);
    atomicAdd(hp + 1, v.y);
    atomicAdd(hp + 2, v.z);
    atomicAdd(hp + 3, v.w);
}

__global__ void relu_copy(const float* __restrict__ in, float* __restrict__ out, long n) {
    const long i = (long)blockIdx.x * blockDim.x + threadIdx.x;
    if (i >= n) return;
    const float v = in[i];
    out[i] = v > 0.0f ? v : 0.0f;
}

extern "C" void kernel_launch(void* const* d_in, const int* in_sizes, int n_in,
                              void* d_out, int out_size, void* d_ws, size_t ws_size,
                              hipStream_t stream) {
    const int D_IN = 128, D_H = 96, D_OUT = 32;
    const float* x0      = (const float*)d_in[0];
    const long long* ei  = (const long long*)d_in[1];
    const float* W[3]  = {(const float*)d_in[2],  (const float*)d_in[6],  (const float*)d_in[10]};
    const float* b[3]  = {(const float*)d_in[3],  (const float*)d_in[7],  (const float*)d_in[11]};
    const float* Ws[3] = {(const float*)d_in[4],  (const float*)d_in[8],  (const float*)d_in[12]};
    const float* bs[3] = {(const float*)d_in[5],  (const float*)d_in[9],  (const float*)d_in[13]};
    const float* mW1 = (const float*)d_in[14];
    const float* mb1 = (const float*)d_in[15];
    const float* mW2 = (const float*)d_in[16];
    const float* mb2 = (const float*)d_in[17];

    const int N = in_sizes[0] / D_IN;
    const int E = in_sizes[1] / 2;

    // ---- workspace layout ----
    float* T   = (float*)d_ws;                 // [N, 96] messages / t1
    float* H   = T + (size_t)N * D_H;          // [N, 96] conv accumulator
    float* X   = H + (size_t)N * D_H;          // [N, 96] relu'd activation
    // packed bf16 weights, 32B-aligned (X end is already 32B aligned for even N)
    __bf16* pk = (__bf16*)(X + (size_t)N * D_H);
    const size_t frag128_6 = (size_t)(128 / 32) * 6 * 32 * 16;  // K=128, NT=6
    const size_t frag96_6  = (size_t)(96  / 32) * 6 * 32 * 16;  // K=96,  NT=6
    const size_t frag96_2  = (size_t)(96  / 32) * 2 * 32 * 16;  // K=96,  NT=2
    __bf16* pW[3]  = {pk, pk + frag128_6, pk + frag128_6 + frag96_6};
    __bf16* pWs[3] = {pW[2] + frag96_6, pW[2] + frag96_6 + frag128_6,
                      pW[2] + frag96_6 + frag128_6 + frag96_6};
    __bf16* pmW1 = pWs[2] + frag96_6;
    __bf16* pmW2 = pmW1 + frag96_6;

    float* emb = (float*)d_out;                // [N, 96] layer-2 pre-ReLU
    float* out = emb + (size_t)N * D_H;        // [N, 32]

    const int mtiles = N / 16;                             // 3125 (exact)
    const dim3 gblk(128);                                  // 4 waves / block
    const dim3 ggrd((mtiles + 3) / 4);
    const long sthreads = (long)E * (D_H / 4);
    const dim3 sgrd((unsigned)((sthreads + 255) / 256)), sblk(256);
    const long rn = (long)N * D_H;
    const dim3 rgrd((unsigned)((rn + 255) / 256)), rblk(256);
    const dim3 pblk(256);
    const dim3 pgrd6_128((128 / 32) * 6 * 32 / 256 + 1);
    const dim3 pgrd6_96((96 / 32) * 6 * 32 / 256 + 1);
    const dim3 pgrd2_96((96 / 32) * 2 * 32 / 256 + 1);

    // ---- pack all weight matrices into WMMA fragment layout (bf16) ----
    pack_w_bf16<<<pgrd6_128, pblk, 0, stream>>>(W[0],  pW[0],  D_IN, 6);
    pack_w_bf16<<<pgrd6_128, pblk, 0, stream>>>(Ws[0], pWs[0], D_IN, 6);
    pack_w_bf16<<<pgrd6_96,  pblk, 0, stream>>>(W[1],  pW[1],  D_H,  6);
    pack_w_bf16<<<pgrd6_96,  pblk, 0, stream>>>(Ws[1], pWs[1], D_H,  6);
    pack_w_bf16<<<pgrd6_96,  pblk, 0, stream>>>(W[2],  pW[2],  D_H,  6);
    pack_w_bf16<<<pgrd6_96,  pblk, 0, stream>>>(Ws[2], pWs[2], D_H,  6);
    pack_w_bf16<<<pgrd6_96,  pblk, 0, stream>>>(mW1,   pmW1,   D_H,  6);
    pack_w_bf16<<<pgrd2_96,  pblk, 0, stream>>>(mW2,   pmW2,   D_H,  2);

    // ---- layer 0 (K = 128) ----
    gemm_packed<6><<<ggrd, gblk, 0, stream>>>(x0, pW[0],  b[0],  T, N, D_IN, 0);
    gemm_packed<6><<<ggrd, gblk, 0, stream>>>(x0, pWs[0], bs[0], H, N, D_IN, 0);
    scatter_edges<<<sgrd, sblk, 0, stream>>>(T, ei, H, E, D_H);
    relu_copy<<<rgrd, rblk, 0, stream>>>(H, X, rn);

    // ---- layer 1 (K = 96) ----
    gemm_packed<6><<<ggrd, gblk, 0, stream>>>(X, pW[1],  b[1],  T, N, D_H, 0);
    gemm_packed<6><<<ggrd, gblk, 0, stream>>>(X, pWs[1], bs[1], H, N, D_H, 0);
    scatter_edges<<<sgrd, sblk, 0, stream>>>(T, ei, H, E, D_H);
    relu_copy<<<rgrd, rblk, 0, stream>>>(H, X, rn);

    // ---- layer 2: self-path GEMM writes emb (d_out) directly ----
    gemm_packed<6><<<ggrd, gblk, 0, stream>>>(X, pW[2],  b[2],  T,   N, D_H, 0);
    gemm_packed<6><<<ggrd, gblk, 0, stream>>>(X, pWs[2], bs[2], emb, N, D_H, 0);
    scatter_edges<<<sgrd, sblk, 0, stream>>>(T, ei, emb, E, D_H);
    relu_copy<<<rgrd, rblk, 0, stream>>>(emb, X, rn);

    // ---- post_mp: relu fused into first GEMM ----
    gemm_packed<6><<<ggrd, gblk, 0, stream>>>(X, pmW1, mb1, T,   N, D_H, 1);
    gemm_packed<2><<<ggrd, gblk, 0, stream>>>(T, pmW2, mb2, out, N, D_H, 0);
}
